// Hawk_24180665877249
// MI455X (gfx1250) — matrix operand order repaired and checked by reading, not compiled, not run on hardware
//
#include <hip/hip_runtime.h>
#include <math.h>

// ---------------------------------------------------------------------------
// Hawk / RG-LRU block for MI455X (gfx1250, wave32, WMMA).
//   cvt(x,Win,Wg,Wout -> bf16)
//   GEMM1: u = x @ Win^T                       (bf16 WMMA, bf16 out)
//   conv : causal depthwise K=4 on xh half     (bf16 out)
//   GEMM2: g = xh @ Wg^T + b_gates             (bf16 WMMA, f32 out)
//   scan : chunked RG-LRU scan + gelu(gate)*h  (bf16 out)
//   GEMM3: out = y @ Wout^T                    (bf16 WMMA, f32 out)
// GEMMs are double-buffered through LDS with async global->LDS DMA
// (GLOBAL_LOAD_ASYNC_TO_LDS_B128 / ASYNCcnt) when the builtin exists.
// ---------------------------------------------------------------------------

#define NBATCH 8
#define TLEN   2048
#define MT     (NBATCH * TLEN)   // 16384 token rows
#define DIMC   1024
#define HID    1536
#define G2     (2 * HID)         // 3072
#define KCONV  4
#define NCH    32                // scan chunks
#define CHT    (TLEN / NCH)      // 64 steps per chunk

typedef __attribute__((ext_vector_type(16))) __bf16 bf16x16;
typedef __attribute__((ext_vector_type(8)))  __bf16 bf16x8;
typedef __attribute__((ext_vector_type(8)))  float  f32x8;

#if defined(__has_builtin)
#if __has_builtin(__builtin_amdgcn_global_load_async_to_lds_b128)
#define HAVE_ASYNC_LDS 1
#endif
#endif
#ifndef HAVE_ASYNC_LDS
#define HAVE_ASYNC_LDS 0
#endif

// exact parameter types of the async builtin: int4 in AS(1)/AS(3)
typedef int v4i __attribute__((vector_size(16)));
typedef __attribute__((address_space(1))) v4i av4i_g;
typedef __attribute__((address_space(3))) v4i av4i_l;

__device__ __forceinline__ void async_cp_b128(const unsigned short* g, unsigned short* l) {
#if HAVE_ASYNC_LDS
  // global->AS1 keeps the flat address; LDS offset is the low 32 bits of the
  // generic address of a __shared__ object.
  __builtin_amdgcn_global_load_async_to_lds_b128(
      (av4i_g*)(unsigned long long)(const void*)g,
      (av4i_l*)(unsigned int)(unsigned long long)(void*)l, 0, 0);
#else
  *(uint4*)l = *(const uint4*)g;  // sync fallback
#endif
}
__device__ __forceinline__ void wait_async0() {
#if HAVE_ASYNC_LDS
#if __has_builtin(__builtin_amdgcn_s_wait_asynccnt)
  __builtin_amdgcn_s_wait_asynccnt(0);
#else
  asm volatile("s_wait_asynccnt 0x0" ::: "memory");
#endif
#endif
}

__device__ __forceinline__ unsigned short f2bf(float f) {
  unsigned int u = __float_as_uint(f);
  u += 0x7FFFu + ((u >> 16) & 1u);          // round-to-nearest-even
  return (unsigned short)(u >> 16);
}
__device__ __forceinline__ float bf2f(unsigned short h) {
  return __uint_as_float(((unsigned int)h) << 16);
}
__device__ __forceinline__ float sigmoidf_(float x) { return 1.0f / (1.0f + __expf(-x)); }
__device__ __forceinline__ float geluf_(float x) {
  return 0.5f * x * (1.0f + erff(x * 0.7071067811865475f));
}

// ---------------------------------------------------------------------------
// fp32 -> bf16 conversion
// ---------------------------------------------------------------------------
__global__ void hawk_cvt_bf16(const float* __restrict__ in,
                              unsigned short* __restrict__ out, long long n) {
  long long i = (long long)blockIdx.x * blockDim.x + threadIdx.x;
  if (i < n) out[i] = f2bf(in[i]);
}

// ---------------------------------------------------------------------------
// bf16 WMMA GEMM: C[M,N] = A[M,K] * B[N,K]^T (+bias), A/B bf16 row-major.
// Block: 256 threads (8 waves), tile 128x128, K-step 32.
// Double-buffered LDS, async global->LDS staging overlapped with WMMA.
// Wave grid 2(M) x 4(N); each wave: 4 M-frags x 2 N-frags of 16x16.
// ---------------------------------------------------------------------------
#define LDSPAD 40  // row stride in halves (32 data + 8 pad)

template <bool BIAS, bool OUT_BF16>
__global__ __launch_bounds__(256)
void hawk_gemm_bf16(const unsigned short* __restrict__ A,
                    const unsigned short* __restrict__ B,
                    const float* __restrict__ bias,
                    void* __restrict__ Cout,
                    int M, int N, int Kt) {
  __shared__ unsigned short lsA[2][128 * LDSPAD];
  __shared__ unsigned short lsB[2][128 * LDSPAD];

  const int tid  = threadIdx.x;
  const int lane = tid & 31;
  const int wave = tid >> 5;       // 0..7
  const int wm   = wave >> 2;      // 0..1 -> 64-row slab
  const int wn   = wave & 3;       // 0..3 -> 32-col slab
  const int lrow = lane & 15;
  const int hi   = lane >> 4;      // lane group select

  const int mBase = blockIdx.y * 128;
  const int nBase = blockIdx.x * 128;

  f32x8 acc[4][2];
#pragma unroll
  for (int i = 0; i < 4; ++i)
#pragma unroll
    for (int j = 0; j < 2; ++j) {
      f32x8 z = {0.f, 0.f, 0.f, 0.f, 0.f, 0.f, 0.f, 0.f};
      acc[i][j] = z;
    }

  // cooperative staging coords: 4 b128 transfers per thread per K-step
  const int cr = tid >> 2;          // row 0..63 (and +64)
  const int cc = (tid & 3) * 8;     // half offset within 32-wide K slab
  const unsigned short* gA0 = A + (size_t)(mBase + cr) * Kt + cc;
  const unsigned short* gA1 = A + (size_t)(mBase + cr + 64) * Kt + cc;
  const unsigned short* gB0 = B + (size_t)(nBase + cr) * Kt + cc;
  const unsigned short* gB1 = B + (size_t)(nBase + cr + 64) * Kt + cc;

  auto stage = [&](int buf, int kBase) {
    async_cp_b128(gA0 + kBase, &lsA[buf][cr * LDSPAD + cc]);
    async_cp_b128(gA1 + kBase, &lsA[buf][(cr + 64) * LDSPAD + cc]);
    async_cp_b128(gB0 + kBase, &lsB[buf][cr * LDSPAD + cc]);
    async_cp_b128(gB1 + kBase, &lsB[buf][(cr + 64) * LDSPAD + cc]);
  };

  const int nk = Kt >> 5;
  stage(0, 0);

  for (int kt = 0; kt < nk; ++kt) {
    const int cur = kt & 1, nxt = cur ^ 1;
    wait_async0();       // my DMA for buffer `cur` has landed in LDS
    __syncthreads();     // everyone's `cur` landed; everyone done reading `nxt`
    if (kt + 1 < nk) stage(nxt, (kt + 1) << 5);  // DMA flies under the WMMAs

    // A fragments: 16x32 bf16; lane holds row (base+lrow),
    // halves [0..7] = K 8*hi.., halves [8..15] = K 16+8*hi..
    bf16x16 afr[4];
#pragma unroll
    for (int f = 0; f < 4; ++f) {
      const int r = wm * 64 + f * 16 + lrow;
      bf16x8 lo  = *(const bf16x8*)(&lsA[cur][r * LDSPAD + hi * 8]);
      bf16x8 hi8 = *(const bf16x8*)(&lsA[cur][r * LDSPAD + hi * 8 + 16]);
      afr[f] = __builtin_shufflevector(lo, hi8, 0, 1, 2, 3, 4, 5, 6, 7,
                                       8, 9, 10, 11, 12, 13, 14, 15);
    }
    // B fragments: 32x16 bf16; lane holds weight row (base+lrow),
    // halves [0..7] = K 8*hi.., halves [8..15] = K 16+8*hi..
    bf16x16 bfr[2];
#pragma unroll
    for (int f = 0; f < 2; ++f) {
      const int r = wn * 32 + f * 16 + lrow;
      bf16x8 lo  = *(const bf16x8*)(&lsB[cur][r * LDSPAD + hi * 8]);
      bf16x8 hi8 = *(const bf16x8*)(&lsB[cur][r * LDSPAD + hi * 8 + 16]);
      bfr[f] = __builtin_shufflevector(lo, hi8, 0, 1, 2, 3, 4, 5, 6, 7,
                                       8, 9, 10, 11, 12, 13, 14, 15);
    }
#pragma unroll
    for (int i = 0; i < 4; ++i)
#pragma unroll
      for (int j = 0; j < 2; ++j)
        acc[i][j] = __builtin_amdgcn_wmma_f32_16x16x32_bf16(
            false, afr[i], false, bfr[j], (short)0, acc[i][j], false, false);
  }

  // Epilogue: D element r -> row (r + 8*hi), col (lane&15).
#pragma unroll
  for (int i = 0; i < 4; ++i)
#pragma unroll
    for (int j = 0; j < 2; ++j) {
      const int col = nBase + wn * 32 + j * 16 + lrow;
#pragma unroll
      for (int r = 0; r < 8; ++r) {
        const int m = mBase + wm * 64 + i * 16 + hi * 8 + r;
        float v = acc[i][j][r];
        if constexpr (BIAS) v += bias[col];
        if constexpr (OUT_BF16)
          ((unsigned short*)Cout)[(size_t)m * N + col] = f2bf(v);
        else
          ((float*)Cout)[(size_t)m * N + col] = v;
      }
    }
}

// ---------------------------------------------------------------------------
// Causal depthwise conv (K=4): xh[t] = b[h] + sum_k w[h,k]*u_xh[t-3+k]
// u holds [gate | xh_pre] concatenated as [M, 2H] bf16; xh half at col H+h.
// ---------------------------------------------------------------------------
__global__ void hawk_conv(const unsigned short* __restrict__ u,
                          const float* __restrict__ cw,
                          const float* __restrict__ cb,
                          unsigned short* __restrict__ xh) {
  const long long idx = (long long)blockIdx.x * blockDim.x + threadIdx.x;
  if (idx >= (long long)MT * HID) return;
  const int h = (int)(idx % HID);
  const int t = (int)((idx / HID) % TLEN);
  const int n = (int)(idx / ((long long)HID * TLEN));
  float acc = cb[h];
#pragma unroll
  for (int k = 0; k < KCONV; ++k) {
    const int ts = t + k - (KCONV - 1);
    if (ts >= 0) {
      const size_t row = (size_t)n * TLEN + ts;
      acc += cw[h * KCONV + k] * bf2f(u[row * G2 + HID + h]);
    }
  }
  xh[(size_t)((size_t)n * TLEN + t) * HID + h] = f2bf(acc);
}

// ---------------------------------------------------------------------------
// Chunked RG-LRU scan. Pass A: per-chunk (prod alpha, local state).
// ---------------------------------------------------------------------------
__global__ void hawk_scanA(const float* __restrict__ g,
                           const unsigned short* __restrict__ xh,
                           const float* __restrict__ fb,
                           float* __restrict__ Pc, float* __restrict__ Sc) {
  const long long idx = (long long)blockIdx.x * blockDim.x + threadIdx.x;
  if (idx >= (long long)NBATCH * NCH * HID) return;
  const int h = (int)(idx % HID);
  const int c = (int)((idx / HID) % NCH);
  const int n = (int)(idx / ((long long)HID * NCH));
  const float kmul = -8.0f * log1pf(__expf(fb[h]));  // ALPHA_LOG_SCALE*softplus
  float P = 1.0f, S = 0.0f;
  for (int t = c * CHT; t < (c + 1) * CHT; ++t) {
    const size_t row = (size_t)n * TLEN + t;
    const float f  = g[row * G2 + h];
    const float ip = g[row * G2 + HID + h];
    const float al = __expf(kmul * sigmoidf_(f));
    const float xv = bf2f(xh[row * HID + h]);
    const float beta = sqrtf(1.0f - al * al + 1e-6f);
    const float xs = beta * sigmoidf_(ip) * xv;
    P *= al;
    S = al * S + xs;
  }
  Pc[((size_t)n * NCH + c) * HID + h] = P;
  Sc[((size_t)n * NCH + c) * HID + h] = S;
}

// Pass B: serial scan over the 32 chunk summaries -> carry-in per chunk.
__global__ void hawk_scanB(const float* __restrict__ Pc,
                           const float* __restrict__ Sc,
                           float* __restrict__ Hin) {
  const long long idx = (long long)blockIdx.x * blockDim.x + threadIdx.x;
  if (idx >= (long long)NBATCH * HID) return;
  const int h = (int)(idx % HID);
  const int n = (int)(idx / HID);
  float run = 0.0f;
#pragma unroll
  for (int c = 0; c < NCH; ++c) {
    const size_t o = ((size_t)n * NCH + c) * HID + h;
    Hin[o] = run;
    run = Pc[o] * run + Sc[o];
  }
}

// Pass C: re-run chunk with carry-in, fuse y = gelu(gate) * h -> bf16.
__global__ void hawk_scanC(const float* __restrict__ g,
                           const unsigned short* __restrict__ xh,
                           const unsigned short* __restrict__ u,
                           const float* __restrict__ fb,
                           const float* __restrict__ Hin,
                           unsigned short* __restrict__ y) {
  const long long idx = (long long)blockIdx.x * blockDim.x + threadIdx.x;
  if (idx >= (long long)NBATCH * NCH * HID) return;
  const int h = (int)(idx % HID);
  const int c = (int)((idx / HID) % NCH);
  const int n = (int)(idx / ((long long)HID * NCH));
  const float kmul = -8.0f * log1pf(__expf(fb[h]));
  float hs = Hin[((size_t)n * NCH + c) * HID + h];
  for (int t = c * CHT; t < (c + 1) * CHT; ++t) {
    const size_t row = (size_t)n * TLEN + t;
    const float f  = g[row * G2 + h];
    const float ip = g[row * G2 + HID + h];
    const float al = __expf(kmul * sigmoidf_(f));
    const float xv = bf2f(xh[row * HID + h]);
    const float beta = sqrtf(1.0f - al * al + 1e-6f);
    hs = al * hs + beta * sigmoidf_(ip) * xv;
    const float gate = bf2f(u[row * G2 + h]);
    y[row * HID + h] = f2bf(geluf_(gate) * hs);
  }
}

// ---------------------------------------------------------------------------
extern "C" void kernel_launch(void* const* d_in, const int* in_sizes, int n_in,
                              void* d_out, int out_size, void* d_ws, size_t ws_size,
                              hipStream_t stream) {
  (void)in_sizes; (void)n_in; (void)out_size;
  const float* x     = (const float*)d_in[0];
  const float* Win   = (const float*)d_in[1];
  const float* convw = (const float*)d_in[2];
  const float* convb = (const float*)d_in[3];
  const float* Wg    = (const float*)d_in[4];
  const float* bg    = (const float*)d_in[5];
  const float* fb    = (const float*)d_in[6];
  const float* Wout  = (const float*)d_in[7];

  char* ws = (char*)d_ws;
  size_t off = 0;
  auto take = [&](size_t bytes) -> void* {
    void* p = (void*)(ws + off);
    off += (bytes + 255) & ~(size_t)255;
    return p;
  };
  unsigned short* x_bf    = (unsigned short*)take((size_t)MT * DIMC * 2);
  unsigned short* Win_bf  = (unsigned short*)take((size_t)G2 * DIMC * 2);
  unsigned short* Wg_bf   = (unsigned short*)take((size_t)G2 * HID * 2);
  unsigned short* Wout_bf = (unsigned short*)take((size_t)DIMC * HID * 2);
  unsigned short* u_bf    = (unsigned short*)take((size_t)MT * G2 * 2);
  unsigned short* xh_bf   = (unsigned short*)take((size_t)MT * HID * 2);
  float*          g_f32   = (float*)take((size_t)MT * G2 * 4);
  unsigned short* y_bf    = (unsigned short*)take((size_t)MT * HID * 2);
  float*          Pc      = (float*)take((size_t)NBATCH * NCH * HID * 4);
  float*          Sc      = (float*)take((size_t)NBATCH * NCH * HID * 4);
  float*          Hin     = (float*)take((size_t)NBATCH * NCH * HID * 4);
  if (off > ws_size) return;  // workspace too small: deterministic no-op

  const int TPB = 256;
  auto blocks = [](long long n) { return (unsigned)((n + 255) / 256); };

  // fp32 -> bf16 operand conversions
  hawk_cvt_bf16<<<blocks((long long)MT * DIMC), TPB, 0, stream>>>(x, x_bf, (long long)MT * DIMC);
  hawk_cvt_bf16<<<blocks((long long)G2 * DIMC), TPB, 0, stream>>>(Win, Win_bf, (long long)G2 * DIMC);
  hawk_cvt_bf16<<<blocks((long long)G2 * HID), TPB, 0, stream>>>(Wg, Wg_bf, (long long)G2 * HID);
  hawk_cvt_bf16<<<blocks((long long)DIMC * HID), TPB, 0, stream>>>(Wout, Wout_bf, (long long)DIMC * HID);

  // GEMM1: u = x @ Win^T  [MT, G2]
  {
    dim3 grid(G2 / 128, MT / 128);
    hawk_gemm_bf16<false, true><<<grid, TPB, 0, stream>>>(
        x_bf, Win_bf, nullptr, (void*)u_bf, MT, G2, DIMC);
  }
  // causal depthwise conv on xh half
  hawk_conv<<<blocks((long long)MT * HID), TPB, 0, stream>>>(u_bf, convw, convb, xh_bf);

  // GEMM2: g = xh @ Wg^T + b_gates  [MT, G2] fp32
  {
    dim3 grid(G2 / 128, MT / 128);
    hawk_gemm_bf16<true, false><<<grid, TPB, 0, stream>>>(
        xh_bf, Wg_bf, bg, (void*)g_f32, MT, G2, HID);
  }

  // chunked RG-LRU scan, fused gelu(gate)*h epilogue
  hawk_scanA<<<blocks((long long)NBATCH * NCH * HID), TPB, 0, stream>>>(g_f32, xh_bf, fb, Pc, Sc);
  hawk_scanB<<<blocks((long long)NBATCH * HID), TPB, 0, stream>>>(Pc, Sc, Hin);
  hawk_scanC<<<blocks((long long)NBATCH * NCH * HID), TPB, 0, stream>>>(g_f32, xh_bf, u_bf, fb, Hin, y_bf);

  // GEMM3: out = y @ Wout^T  [MT, DIMC] fp32
  {
    dim3 grid(DIMC / 128, MT / 128);
    hawk_gemm_bf16<false, false><<<grid, TPB, 0, stream>>>(
        y_bf, Wout_bf, nullptr, d_out, MT, DIMC, HID);
  }
}
